// GPT_69707319214145
// MI455X (gfx1250) — matrix operand order repaired and checked by reading, not compiled, not run on hardware
//
#include <hip/hip_runtime.h>
#include <math.h>

// ---------------- problem constants ----------------
#define Bb   2
#define Tt   2048
#define Vv   32768
#define Dd   512
#define Ll   8
#define Hh   4
#define HDd  128
#define GATE_CH 32
#define SHORT_W 128
#define Mtok (Bb*Tt)   // 4096

// ---------------- CDNA5 async-copy availability ----------------
#if defined(__AMDGCN__) && __has_builtin(__builtin_amdgcn_global_load_async_to_lds_b128)
#define HAS_ASYNC_LDS 1
#else
#define HAS_ASYNC_LDS 0
#endif

#if HAS_ASYNC_LDS
typedef int vi4 __attribute__((vector_size(4 * sizeof(int))));
__device__ __forceinline__ void async_copy_b128(const void* gsrc, void* ldst) {
    __builtin_amdgcn_global_load_async_to_lds_b128(
        (__attribute__((address_space(1))) vi4*)gsrc,
        (__attribute__((address_space(3))) vi4*)ldst, 0, 0);
}
__device__ __forceinline__ void async_wait0() {
#if __has_builtin(__builtin_amdgcn_s_wait_asynccnt)
    __builtin_amdgcn_s_wait_asynccnt(0);
#else
    asm volatile("s_wait_asynccnt 0x0" ::: "memory");
#endif
}
#endif

// ---------------- types ----------------
typedef __attribute__((ext_vector_type(16))) __bf16        v16bf;
typedef __attribute__((ext_vector_type(8)))  float         v8f;
typedef __attribute__((ext_vector_type(4)))  unsigned int  u32x4;
typedef __attribute__((ext_vector_type(4)))  float         f32x4;
typedef __attribute__((ext_vector_type(2)))  float         f32x2;

union Frag16 { u32x4 u[2]; v16bf bf; };   // 32 bytes: one WMMA 16-bit A/B operand

__device__ __forceinline__ unsigned short f2bf(float f) {
    union { float f; unsigned int u; } c; c.f = f;
    unsigned int u = c.u;
    unsigned int r = u + 0x7FFFu + ((u >> 16) & 1u);   // round-to-nearest-even
    return (unsigned short)(r >> 16);
}
__device__ __forceinline__ float bf2f(unsigned short b) {
    union { float f; unsigned int u; } c; c.u = ((unsigned int)b) << 16;
    return c.f;
}

// =====================================================================
// embed: x[row] = rmsnorm(wte[idx[row]])      (row = b*T+t, f32 out)
// =====================================================================
__global__ __launch_bounds__(256) void embed_kernel(
    const int* __restrict__ idx, const float* __restrict__ wte,
    float* __restrict__ x)
{
    int row = blockIdx.x, tid = threadIdx.x;
    int tok = idx[row];
    const float* wr = wte + (size_t)tok * Dd;
    f32x2 v = ((const f32x2*)wr)[tid];
    __shared__ float red[256];
    red[tid] = v.x * v.x + v.y * v.y;
    __syncthreads();
    for (int s = 128; s > 0; s >>= 1) { if (tid < s) red[tid] += red[tid + s]; __syncthreads(); }
    float inv = rsqrtf(red[0] / (float)Dd + 1e-6f);
    float* xr = x + (size_t)row * Dd;
    ((f32x2*)xr)[tid] = f32x2{ v.x * inv, v.y * inv };
}

// =====================================================================
// h = rmsnorm(x) -> bf16 (WMMA A operand layout: row-major [M,D])
// =====================================================================
__global__ __launch_bounds__(256) void rmsnorm_bf16_kernel(
    const float* __restrict__ x, unsigned short* __restrict__ hb)
{
    int row = blockIdx.x, tid = threadIdx.x;
    const float* xr = x + (size_t)row * Dd;
    f32x2 v = ((const f32x2*)xr)[tid];
    __shared__ float red[256];
    red[tid] = v.x * v.x + v.y * v.y;
    __syncthreads();
    for (int s = 128; s > 0; s >>= 1) { if (tid < s) red[tid] += red[tid + s]; __syncthreads(); }
    float inv = rsqrtf(red[0] / (float)Dd + 1e-6f);
    unsigned short* hr = hb + (size_t)row * Dd;
    hr[2 * tid]     = f2bf(v.x * inv);
    hr[2 * tid + 1] = f2bf(v.y * inv);
}

// =====================================================================
// Tiled WMMA GEMM:  C[M,N] = A[M,K](bf16) @ W[K,N](f32, cvt->bf16)
//  MODE 0: Cf = result (f32)
//  MODE 1: Cf += result (residual add, f32)
//  MODE 2: Cb = relu(result)^2 (bf16)
// Tile 128x128, BK=32, 256 threads = 8 waves, wave tile 32(M)x64(N).
// A tile staged via GLOBAL_LOAD_ASYNC_TO_LDS_B128 (ASYNCcnt path) when
// available; W tile needs f32->bf16 convert+transpose so it stays manual.
// =====================================================================
template<int MODE>
__global__ __launch_bounds__(256) void gemm_bf16(
    const unsigned short* __restrict__ A,
    const float* __restrict__ W,
    float* __restrict__ Cf,
    unsigned short* __restrict__ Cb,
    int M, int N, int K)
{
    __shared__ __align__(16) unsigned short As[128 * 40]; // [m][k] pad 32->40
    __shared__ __align__(16) unsigned short Bs[128 * 40]; // [n][k] transposed, pad

    int tid  = threadIdx.x;
    int lane = tid & 31, w = tid >> 5;
    int lrow = lane & 15, lhi = lane >> 4;
    int wm = w & 3, wn = w >> 2;          // 4 x 2 wave grid
    int m0 = blockIdx.y * 128;
    int n0 = blockIdx.x * 128;

    v8f acc[2][4];
    v8f z = { 0.f,0.f,0.f,0.f,0.f,0.f,0.f,0.f };
    #pragma unroll
    for (int i = 0; i < 2; i++)
        #pragma unroll
        for (int j = 0; j < 4; j++) acc[i][j] = z;

    int arow = tid >> 1, acg = tid & 1;   // A stage: 2 thr/row x 16 bf16
    int krow = tid >> 3, ng  = tid & 7;   // W stage: 8 thr/row x 16 f32

    for (int k0 = 0; k0 < K; k0 += 32) {
        // prefetch next W tile (weights dominate HBM traffic)
        if (k0 + 32 < K)
            __builtin_prefetch(&W[(size_t)(k0 + 32 + krow) * N + n0 + 16 * ng], 0, 1);

        __syncthreads();
        // stage A (bf16, row-major): pure copy -> async DMA path
        {
            const unsigned short* gsrc = &A[(size_t)(m0 + arow) * K + k0 + 16 * acg];
            unsigned short*       ldst = &As[arow * 40 + 16 * acg];
#if HAS_ASYNC_LDS
            async_copy_b128(gsrc, ldst);
#else
            const u32x4* ap = (const u32x4*)gsrc;
            u32x4* dp = (u32x4*)ldst;
            dp[0] = ap[0]; dp[1] = ap[1];
#endif
        }
        // stage W (f32 -> bf16, transposed [n][k])
        {
            const f32x4* wp = (const f32x4*)&W[(size_t)(k0 + krow) * N + n0 + 16 * ng];
            f32x4 w0 = wp[0], w1 = wp[1], w2 = wp[2], w3 = wp[3];
            #pragma unroll
            for (int j = 0; j < 4; j++) {
                Bs[(16 * ng + j)      * 40 + krow] = f2bf(w0[j]);
                Bs[(16 * ng + 4 + j)  * 40 + krow] = f2bf(w1[j]);
                Bs[(16 * ng + 8 + j)  * 40 + krow] = f2bf(w2[j]);
                Bs[(16 * ng + 12 + j) * 40 + krow] = f2bf(w3[j]);
            }
        }
#if HAS_ASYNC_LDS
        async_wait0();
#endif
        __syncthreads();

        // A fragments for this wave's two 16-row tiles
        Frag16 fa0, fa1;
        {
            const u32x4* p0 = (const u32x4*)&As[(wm * 32 + lrow) * 40];
            const u32x4* p1 = (const u32x4*)&As[(wm * 32 + 16 + lrow) * 40];
            fa0.u[0] = p0[lhi]; fa0.u[1] = p0[2 + lhi];
            fa1.u[0] = p1[lhi]; fa1.u[1] = p1[2 + lhi];
        }
        #pragma unroll
        for (int ni = 0; ni < 4; ni++) {
            Frag16 fb;
            const u32x4* bp = (const u32x4*)&Bs[(wn * 64 + ni * 16 + lrow) * 40];
            fb.u[0] = bp[lhi]; fb.u[1] = bp[2 + lhi];
            acc[0][ni] = __builtin_amdgcn_wmma_f32_16x16x32_bf16(
                false, fa0.bf, false, fb.bf, (short)0, acc[0][ni], false, false);
            acc[1][ni] = __builtin_amdgcn_wmma_f32_16x16x32_bf16(
                false, fa1.bf, false, fb.bf, (short)0, acc[1][ni], false, false);
        }
    }

    // epilogue: C layout = col lane&15, rows r + 8*(lane>>4)
    #pragma unroll
    for (int mi = 0; mi < 2; mi++) {
        #pragma unroll
        for (int ni = 0; ni < 4; ni++) {
            int gr = m0 + wm * 32 + mi * 16 + 8 * lhi;
            int gc = n0 + wn * 64 + ni * 16 + lrow;
            #pragma unroll
            for (int r = 0; r < 8; r++) {
                float v = acc[mi][ni][r];
                size_t off = (size_t)(gr + r) * N + gc;
                if (MODE == 0) {
                    Cf[off] = v;
                } else if (MODE == 1) {
                    Cf[off] += v;
                } else { // MODE 2: relu^2 -> bf16
                    float t = v > 0.f ? v : 0.f;
                    Cb[off] = f2bf(t * t);
                }
            }
        }
    }
}

// =====================================================================
// qkv_post: RoPE + per-head rmsnorm on q,k ; ve-gate on v ; -> bf16 [B,H,T,HD]
// block = 256 threads: thread = (h<<6)|j, h in 0..3, j in 0..63
// =====================================================================
__global__ __launch_bounds__(256) void qkv_post_kernel(
    const float* __restrict__ qf, const float* __restrict__ kf,
    const float* __restrict__ vf,
    const unsigned short* __restrict__ hb, const int* __restrict__ idx,
    const float* __restrict__ ve_table,   // V x D, or nullptr
    const float* __restrict__ gate_w,     // GATE_CH x H, or nullptr
    unsigned short* __restrict__ Qb, unsigned short* __restrict__ Kb,
    unsigned short* __restrict__ Vb)
{
    int row = blockIdx.x;             // token = b*T + t
    int b = row / Tt, t = row % Tt;
    int tid = threadIdx.x;
    int h = tid >> 6, j = tid & 63;

    float ang = (float)t * __powf(10000.f, -(float)j / 64.f);
    float c = __cosf(ang), s = __sinf(ang);

    __shared__ float red[256];
    size_t hofs = ((size_t)(b * Hh + h) * Tt + t) * HDd;

    // ---- q ----
    {
        const float* qr = qf + (size_t)row * Dd + h * HDd;
        float x1 = qr[j], x2 = qr[64 + j];
        float o1 = x1 * c + x2 * s, o2 = -x1 * s + x2 * c;
        red[tid] = o1 * o1 + o2 * o2;
        __syncthreads();
        for (int st = 32; st > 0; st >>= 1) { if (j < st) red[tid] += red[tid + st]; __syncthreads(); }
        float inv = rsqrtf(red[h * 64] / (float)HDd + 1e-6f);
        Qb[hofs + j]      = f2bf(o1 * inv);
        Qb[hofs + 64 + j] = f2bf(o2 * inv);
        __syncthreads();
    }
    // ---- k ----
    {
        const float* kr = kf + (size_t)row * Dd + h * HDd;
        float x1 = kr[j], x2 = kr[64 + j];
        float o1 = x1 * c + x2 * s, o2 = -x1 * s + x2 * c;
        red[tid] = o1 * o1 + o2 * o2;
        __syncthreads();
        for (int st = 32; st > 0; st >>= 1) { if (j < st) red[tid] += red[tid + st]; __syncthreads(); }
        float inv = rsqrtf(red[h * 64] / (float)HDd + 1e-6f);
        Kb[hofs + j]      = f2bf(o1 * inv);
        Kb[hofs + 64 + j] = f2bf(o2 * inv);
        __syncthreads();
    }
    // ---- v (+ value-embedding gate) ----
    {
        const float* vr = vf + (size_t)row * Dd + h * HDd;
        float v1 = vr[j], v2 = vr[64 + j];
        if (ve_table != nullptr) {
            red[tid] = (j < GATE_CH)
                ? bf2f(hb[(size_t)row * Dd + j]) * gate_w[j * Hh + h] : 0.f;
            __syncthreads();
            for (int st = 32; st > 0; st >>= 1) { if (j < st) red[tid] += red[tid + st]; __syncthreads(); }
            float g = 2.f / (1.f + __expf(-red[h * 64]));
            const float* ve = ve_table + (size_t)idx[row] * Dd + h * HDd;
            v1 += g * ve[j];
            v2 += g * ve[64 + j];
        }
        Vb[hofs + j]      = f2bf(v1);
        Vb[hofs + 64 + j] = f2bf(v2);
    }
}

// =====================================================================
// Flash attention, WMMA (bf16 QK^T and PV, f32 softmax).
// 256 threads = 8 waves; wave handles 16 q rows; WG handles 128 q rows.
// Key blocks of 32. Causal + sliding-window mask (dist < W).
// K tile staged via async DMA (pure copy); V tile transposed manually.
// Output bf16 into [B*T, D] with col = h*HD + d (proj GEMM A layout).
// =====================================================================
__global__ __launch_bounds__(256) void attn_kernel(
    const unsigned short* __restrict__ Qb,
    const unsigned short* __restrict__ Kb,
    const unsigned short* __restrict__ Vb,
    unsigned short* __restrict__ Ob,
    int W)
{
    int bh = blockIdx.y;                  // b*H + h
    int b = bh / Hh, h = bh % Hh;
    int qblk = blockIdx.x;
    int tid = threadIdx.x, lane = tid & 31, w = tid >> 5;
    int lrow = lane & 15, lhi = lane >> 4;

    const unsigned short* Qh = Qb + (size_t)bh * Tt * HDd;
    const unsigned short* Kh = Kb + (size_t)bh * Tt * HDd;
    const unsigned short* Vh = Vb + (size_t)bh * Tt * HDd;

    __shared__ __align__(16) unsigned short Kl[32 * 136];   // [key][d]
    __shared__ __align__(16) unsigned short Vt[128 * 40];   // [d][key]
    __shared__ __align__(16) float          Sl[8 * 16 * 33];
    __shared__ __align__(16) unsigned short Pl[8 * 16 * 48];
    __shared__ float ml[8 * 16], ll[8 * 16], al[8 * 16];

    int q0 = qblk * 128 + w * 16;         // wave's first query row

    // Q fragments (16 rows x 128 d, 4 k-steps of 32) straight from global
    Frag16 fq[4];
    {
        const u32x4* qp = (const u32x4*)&Qh[(size_t)(q0 + lrow) * HDd];
        #pragma unroll
        for (int s = 0; s < 4; s++) { fq[s].u[0] = qp[4 * s + lhi]; fq[s].u[1] = qp[4 * s + 2 + lhi]; }
    }
    v8f z = { 0.f,0.f,0.f,0.f,0.f,0.f,0.f,0.f };
    v8f o[8];
    #pragma unroll
    for (int i = 0; i < 8; i++) o[i] = z;
    if (lane < 16) { ml[w * 16 + lane] = -1e30f; ll[w * 16 + lane] = 0.f; }

    int kstart = qblk * 128 - (W - 1); if (kstart < 0) kstart = 0;
    kstart &= ~31;
    int kend = qblk * 128 + 127;
    const float scale = 0.088388347648318447f;   // 1/sqrt(128)

    for (int j0 = kstart; j0 <= kend; j0 += 32) {
        __syncthreads();
        // stage K row-major (async DMA), V transposed (manual)
        {
            int r = tid >> 3, g = tid & 7;
            const unsigned short* ksrc = &Kh[(size_t)(j0 + r) * HDd + 16 * g];
            unsigned short*       kdst = &Kl[r * 136 + 16 * g];
#if HAS_ASYNC_LDS
            async_copy_b128(ksrc, kdst);
#else
            const u32x4* kp = (const u32x4*)ksrc;
            u32x4* kd = (u32x4*)kdst;
            kd[0] = kp[0]; kd[1] = kp[1];
#endif
            const u32x4* vp = (const u32x4*)&Vh[(size_t)(j0 + r) * HDd + 16 * g];
            u32x4 va = vp[0], vb2 = vp[1];
            #pragma unroll
            for (int jj = 0; jj < 8; jj++) {
                Vt[(16 * g + jj)     * 40 + r] = (unsigned short)(va[jj >> 1]  >> ((jj & 1) * 16));
                Vt[(16 * g + 8 + jj) * 40 + r] = (unsigned short)(vb2[jj >> 1] >> ((jj & 1) * 16));
            }
        }
#if HAS_ASYNC_LDS
        async_wait0();
#endif
        __syncthreads();

        // S = Q @ K^T  (two 16-key n-tiles)
        #pragma unroll
        for (int nt = 0; nt < 2; nt++) {
            v8f s8 = z;
            const u32x4* kbp = (const u32x4*)&Kl[(nt * 16 + lrow) * 136];
            #pragma unroll
            for (int s = 0; s < 4; s++) {
                Frag16 fb;
                fb.u[0] = kbp[4 * s + lhi];
                fb.u[1] = kbp[4 * s + 2 + lhi];
                s8 = __builtin_amdgcn_wmma_f32_16x16x32_bf16(
                    false, fq[s].bf, false, fb.bf, (short)0, s8, false, false);
            }
            #pragma unroll
            for (int r = 0; r < 8; r++)
                Sl[w * 528 + (r + 8 * lhi) * 33 + nt * 16 + lrow] = s8[r];
        }

        // online softmax: one lane per query row
        if (lane < 16) {
            int r = lane;
            int qi = q0 + r;
            float mo = ml[w * 16 + r], lo = ll[w * 16 + r];
            float mx = mo;
            float sc[32];
            #pragma unroll
            for (int c = 0; c < 32; c++) {
                int j = j0 + c;
                bool valid = (j <= qi) && (qi - j < W);
                float v = valid ? Sl[w * 528 + r * 33 + c] * scale : -1e30f;
                sc[c] = v;
                if (v > mx) mx = v;
            }
            float alpha = __expf(mo - mx);
            float lsum = 0.f;
            #pragma unroll
            for (int c = 0; c < 32; c++) {
                float p = (sc[c] > -9.0e29f) ? __expf(sc[c] - mx) : 0.f;
                lsum += p;
                Pl[w * 768 + r * 48 + c] = f2bf(p);
            }
            ml[w * 16 + r] = mx;
            ll[w * 16 + r] = lo * alpha + lsum;
            al[w * 16 + r] = alpha;
        }

        // O = O*alpha + P @ V
        float a8[8];
        #pragma unroll
        for (int r = 0; r < 8; r++) a8[r] = al[w * 16 + r + 8 * lhi];
        #pragma unroll
        for (int nt = 0; nt < 8; nt++)
            #pragma unroll
            for (int r = 0; r < 8; r++) o[nt][r] *= a8[r];

        Frag16 fp;
        const u32x4* pp = (const u32x4*)&Pl[w * 768 + lrow * 48];
        fp.u[0] = pp[lhi]; fp.u[1] = pp[2 + lhi];

        #pragma unroll
        for (int nt = 0; nt < 8; nt++) {
            Frag16 fv;
            const u32x4* vbp = (const u32x4*)&Vt[(nt * 16 + lrow) * 40];
            fv.u[0] = vbp[lhi]; fv.u[1] = vbp[2 + lhi];
            o[nt] = __builtin_amdgcn_wmma_f32_16x16x32_bf16(
                false, fp.bf, false, fv.bf, (short)0, o[nt], false, false);
        }
    }

    // finalize: divide by l, store bf16 into [B*T, D]
    float inv8[8];
    #pragma unroll
    for (int r = 0; r < 8; r++) inv8[r] = 1.f / ll[w * 16 + r + 8 * lhi];
    #pragma unroll
    for (int nt = 0; nt < 8; nt++) {
        #pragma unroll
        for (int r = 0; r < 8; r++) {
            int q = q0 + r + 8 * lhi;
            size_t off = (size_t)(b * Tt + q) * Dd + h * HDd + nt * 16 + lrow;
            Ob[off] = f2bf(o[nt][r] * inv8[r]);
        }
    }
}

// =====================================================================
// Host orchestration
// =====================================================================
extern "C" void kernel_launch(void* const* d_in, const int* in_sizes, int n_in,
                              void* d_out, int out_size, void* d_ws, size_t ws_size,
                              hipStream_t stream)
{
    (void)in_sizes; (void)n_in; (void)out_size; (void)ws_size;
    const int*   idx        = (const int*)  d_in[0];
    const float* wte        = (const float*)d_in[1];
    const float* ve_tables  = (const float*)d_in[2];
    const float* wq         = (const float*)d_in[3];
    const float* wk         = (const float*)d_in[4];
    const float* wv         = (const float*)d_in[5];
    const float* wproj      = (const float*)d_in[6];
    const float* ve_gate_w  = (const float*)d_in[7];
    const float* w_fc       = (const float*)d_in[8];
    const float* w_fc_proj  = (const float*)d_in[9];
    const float* lm_head    = (const float*)d_in[10];
    float* out = (float*)d_out;

    char* p = (char*)d_ws;
    auto alloc = [&](size_t bytes) -> void* {
        void* r = (void*)p;
        p += (bytes + 255) & ~(size_t)255;
        return r;
    };
    float*          x  = (float*)         alloc((size_t)Mtok * Dd * 4);
    unsigned short* hb = (unsigned short*)alloc((size_t)Mtok * Dd * 2);
    float*          qf = (float*)         alloc((size_t)Mtok * Dd * 4);
    float*          kf = (float*)         alloc((size_t)Mtok * Dd * 4);
    float*          vf = (float*)         alloc((size_t)Mtok * Dd * 4);
    unsigned short* Qb = (unsigned short*)alloc((size_t)Mtok * Dd * 2);
    unsigned short* Kb = (unsigned short*)alloc((size_t)Mtok * Dd * 2);
    unsigned short* Vb = (unsigned short*)alloc((size_t)Mtok * Dd * 2);
    unsigned short* ob = (unsigned short*)alloc((size_t)Mtok * Dd * 2);
    unsigned short* ub = (unsigned short*)alloc((size_t)Mtok * 4 * Dd * 2);

    dim3 blk(256);
    embed_kernel<<<Mtok, blk, 0, stream>>>(idx, wte, x);

    for (int l = 0; l < Ll; l++) {
        rmsnorm_bf16_kernel<<<Mtok, blk, 0, stream>>>(x, hb);

        gemm_bf16<0><<<dim3(Dd / 128, Mtok / 128), blk, 0, stream>>>(
            hb, wq + (size_t)l * Dd * Dd, qf, nullptr, Mtok, Dd, Dd);
        gemm_bf16<0><<<dim3(Dd / 128, Mtok / 128), blk, 0, stream>>>(
            hb, wk + (size_t)l * Dd * Dd, kf, nullptr, Mtok, Dd, Dd);
        gemm_bf16<0><<<dim3(Dd / 128, Mtok / 128), blk, 0, stream>>>(
            hb, wv + (size_t)l * Dd * Dd, vf, nullptr, Mtok, Dd, Dd);

        bool has_ve = ((l % 2) == ((Ll - 1) % 2));
        int  vi     = has_ve ? (l - 1) / 2 : 0;
        qkv_post_kernel<<<Mtok, blk, 0, stream>>>(
            qf, kf, vf, hb, idx,
            has_ve ? (ve_tables + (size_t)vi * Vv * Dd) : nullptr,
            has_ve ? (ve_gate_w + (size_t)vi * GATE_CH * Hh) : nullptr,
            Qb, Kb, Vb);

        int Wwin = ((l % 4) == 3) ? Tt : SHORT_W;   // PATTERN 'SSSL'
        attn_kernel<<<dim3(Tt / 128, Bb * Hh), blk, 0, stream>>>(Qb, Kb, Vb, ob, Wwin);

        gemm_bf16<1><<<dim3(Dd / 128, Mtok / 128), blk, 0, stream>>>(
            ob, wproj + (size_t)l * Dd * Dd, x, nullptr, Mtok, Dd, Dd);

        rmsnorm_bf16_kernel<<<Mtok, blk, 0, stream>>>(x, hb);
        gemm_bf16<2><<<dim3(4 * Dd / 128, Mtok / 128), blk, 0, stream>>>(
            hb, w_fc + (size_t)l * Dd * 4 * Dd, nullptr, ub, Mtok, 4 * Dd, Dd);
        gemm_bf16<1><<<dim3(Dd / 128, Mtok / 128), blk, 0, stream>>>(
            ub, w_fc_proj + (size_t)l * 4 * Dd * Dd, x, nullptr, Mtok, Dd, 4 * Dd);
    }

    rmsnorm_bf16_kernel<<<Mtok, blk, 0, stream>>>(x, hb);
    gemm_bf16<0><<<dim3(Vv / 128, Mtok / 128), blk, 0, stream>>>(
        hb, lm_head, out, nullptr, Mtok, Vv, Dd);
}
